// TransposeDISCOBlock_30391188586833
// MI455X (gfx1250) — compile-verified, matrix operand order checked by reading, not run on hardware
//
#include <hip/hip_runtime.h>
#include <hip/hip_bf16.h>
#include <math.h>

// ---------------- static problem config ----------------
// image: [8, 32, 256, 256] f32 ; weight: [32, 32, 36] f32
// out:   [8, 32, 512, 512] f32
#define NB      8
#define CIN     32
#define COUT    32
#define HIN     256
#define WIN     256
#define HUP     512
#define WUP     512
#define KH      15          // 2*7+1
#define KW      15
#define PH      7
#define PW      7
#define NTAP    (KH*KW)     // 225
#define NBASIS  36          // (6-1)*7 + 1
#define Nb_R    6
#define NPHI    7
#define RCUT    0.0275f     // radius_cutoff / 2 on the upsampled grid
#define EPS_IN  1e-5f
#define NEG_SLOPE 0.2f

typedef __attribute__((ext_vector_type(16))) _Float16 v16h;
typedef __attribute__((ext_vector_type(8)))  float    v8f;

// ---------------- workspace layout ----------------
// [0,       32768)   psi  f32 [36][225]
// [32768,   493568)  filt f16 [225][32][32]  (dword-aligned; read as uint32 pairs)
// [493568,  495616)  stats f32 [256][2]  (mean, rstd)
#define WS_PSI_OFF   0
#define WS_FILT_OFF  32768
#define WS_STAT_OFF  493568

// ================= kernel 1: build psi basis =================
// psi[k][t], t = h*15 + w on a 15x15 grid, normalized so each basis sums to 1.
__global__ void psi_kernel(float* __restrict__ psi) {
    const int k = blockIdx.x;          // 0..35
    const int t = threadIdx.x;         // 0..255 (225 active)
    const float d = 2.0f / (float)HUP; // grid spacing on [-1,1]^2
    const float dr   = RCUT / (float)Nb_R;
    const float dphi = 6.2831853071795864769f / (float)NPHI;

    float val = 0.0f;
    if (t < NTAP) {
        int h = t / KW, w = t - h * KW;
        float y = (float)(h - PH) * d;
        float x = (float)(w - PW) * d;
        float r = sqrtf(x * x + y * y);
        float phi = atan2f(y, x);
        if (phi < 0.0f) phi += 6.2831853071795864769f;
        float ir, iphi, tri_phi;
        if (k == 0) {
            ir = 0.0f; iphi = 0.0f; tri_phi = 1.0f;        // isotropic center bump
        } else {
            ir   = (float)((k - 1) / NPHI + 1) * dr;
            iphi = (float)((k - 1) % NPHI) * dphi;
            float da = fabsf(phi - iphi);
            da = fminf(da, 6.2831853071795864769f - da);   // periodic angular dist
            tri_phi = fmaxf(1.0f - da / dphi, 0.0f);
        }
        float tri_r = fmaxf(1.0f - fabsf(r - ir) / dr, 0.0f);
        val = tri_r * tri_phi * ((r <= RCUT) ? 1.0f : 0.0f);
    }
    __shared__ float sh[256];
    sh[t] = val;
    __syncthreads();
    for (int off = 128; off > 0; off >>= 1) {
        if (t < off) sh[t] += sh[t + off];
        __syncthreads();
    }
    float s = fmaxf(sh[0], 1e-9f);
    if (t < NTAP) psi[k * NTAP + t] = val / s;
}

// ================= kernel 2: fold weight x psi -> dense f16 filter ============
// filt[tap][o][i] = sum_k weight[o][i][k] * psi[k][tap]   (stored f16)
__global__ void fold_kernel(const float* __restrict__ wgt,
                            const float* __restrict__ psi,
                            _Float16* __restrict__ filt) {
    int g = blockIdx.x * 256 + threadIdx.x;
    if (g >= NTAP * COUT * CIN) return;
    int tap = g >> 10;                 // /1024
    int rem = g & 1023;
    int o = rem >> 5, i = rem & 31;
    const float* wp = wgt + (o * CIN + i) * NBASIS;
    float acc = 0.0f;
    #pragma unroll
    for (int k = 0; k < NBASIS; ++k) acc += wp[k] * psi[k * NTAP + tap];
    filt[tap * (COUT * CIN) + o * CIN + i] = (_Float16)acc;
}

// ---------------- bilinear x2 upsample (align_corners=True) at one point ------
__device__ __forceinline__ float upsample_at(const float* __restrict__ plane,
                                             int gy, int gx) {
    const float s = 255.0f / 511.0f;
    float py = (float)gy * s;
    float px = (float)gx * s;
    int iy0 = (int)py; if (iy0 > HIN - 2) iy0 = HIN - 2;
    int ix0 = (int)px; if (ix0 > WIN - 2) ix0 = WIN - 2;
    float fy = py - (float)iy0, fx = px - (float)ix0;
    const float* p = plane + iy0 * WIN + ix0;
    float v00 = p[0], v01 = p[1], v10 = p[WIN], v11 = p[WIN + 1];
    return (v00 * (1.0f - fx) + v01 * fx) * (1.0f - fy)
         + (v10 * (1.0f - fx) + v11 * fx) * fy;
}

// ================= kernel 3: fused upsample + DISCO conv (WMMA) ===============
// One block = one 16x16 output tile of one batch image, all 32 Cout.
// 8 wave32: wave w -> M-half (w&1), N-tiles (w>>1)*4 .. +3 (N = 16 pixels/row).
//
// LDS layout (fragment-native): cell[g][ly][lx] = uint4 holding channel pairs
// c2 = g*4 .. g*4+3 (each dword = 2 x f16 channels) at upsampled point (ly,lx).
// A lane's B fragment = cell[g=khalf] + cell[g=2+khalf]  -> two ds_load_b128,
// landing directly in the contiguous 8-VGPR WMMA operand (no repack movs).
#define TILE_HW 30                       // 16 + 15 - 1 halo
__global__ __launch_bounds__(256) void disco_conv_kernel(
        const float* __restrict__ img,        // [8][32][256][256]
        const unsigned* __restrict__ filt32,  // [225][32][16] dwords (2xf16 each)
        float* __restrict__ out) {            // [8][32][512][512] (pre-norm y)
    __shared__ uint4 lds4[4 * TILE_HW * TILE_HW];   // 57.6 KB

    const int tid = threadIdx.x;
    const int X0 = blockIdx.x * 16;
    const int Y0 = blockIdx.y * 16;
    const int n  = blockIdx.z;
    const float* imgN = img + (size_t)n * CIN * (HIN * WIN);

    // ---- stage: upsample-on-the-fly fill of the 30x30 x 32ch halo tile ----
    unsigned* ldsw = (unsigned*)lds4;
    for (int idx = tid; idx < 16 * (TILE_HW * TILE_HW); idx += 256) {
        int c2  = idx / (TILE_HW * TILE_HW);          // channel pair 0..15
        int rem = idx - c2 * (TILE_HW * TILE_HW);
        int ly = rem / TILE_HW, lx = rem - ly * TILE_HW;
        int gy = Y0 - PH + ly, gx = X0 - PW + lx;
        union { unsigned u; _Float16 h[2]; } pk;
        pk.u = 0u;
        if (gy >= 0 && gy < HUP && gx >= 0 && gx < WUP) {
            const float* p0 = imgN + (2 * c2) * (HIN * WIN);
            pk.h[0] = (_Float16)upsample_at(p0, gy, gx);
            pk.h[1] = (_Float16)upsample_at(p0 + HIN * WIN, gy, gx);
        }
        // cell index: ((g*30 + ly)*30 + lx), dword j = c2 & 3
        ldsw[((((c2 >> 2) * TILE_HW + ly) * TILE_HW + lx) << 2) + (c2 & 3)] = pk.u;
    }
    __syncthreads();

    const int wave  = tid >> 5;
    const int lane  = tid & 31;
    const int m     = wave & 1;       // M-half: Cout 16m..16m+15
    const int ng    = wave >> 1;      // N-group: tiles ng*4 .. ng*4+3
    const int col   = lane & 15;      // A: row M; B: col N; D: col N
    const int khalf = lane >> 4;      // which K-half this lane carries

    // 4 named accumulators -> fixed register tuples across the whole loop
    v8f acc0 = {}, acc1 = {}, acc2 = {}, acc3 = {};

    union Frag { uint4 q[2]; v16h h; };
    // A-side: this lane's filter row base; advances 128 uint4 per tap
    const uint4* fq = (const uint4*)filt32 + (size_t)(m * 16 + col) * 4;
    const int rowbase = ng * 4;

    #pragma unroll 1
    for (int kh = 0; kh < KH; ++kh) {
        // B-side: two channel-group rows for this kh, immediate-offset indexed by kw
        const uint4* br0 = lds4 + ((khalf)     * TILE_HW + rowbase + kh) * TILE_HW + col;
        const uint4* br1 = lds4 + ((2 + khalf) * TILE_HW + rowbase + kh) * TILE_HW + col;
        #pragma unroll 1
        for (int kw = 0; kw < KW; ++kw) {
            Frag a;                      // two global_load_b128 (one s_clause)
            a.q[0] = fq[khalf];
            a.q[1] = fq[2 + khalf];
            fq += COUT * 4;              // next tap (512 dwords)

            Frag b0, b1, b2, b3;         // 8x ds_load_b128, immediate offsets
            b0.q[0] = br0[kw];                b0.q[1] = br1[kw];
            b1.q[0] = br0[kw + TILE_HW];      b1.q[1] = br1[kw + TILE_HW];
            b2.q[0] = br0[kw + 2 * TILE_HW];  b2.q[1] = br1[kw + 2 * TILE_HW];
            b3.q[0] = br0[kw + 3 * TILE_HW];  b3.q[1] = br1[kw + 3 * TILE_HW];

            acc0 = __builtin_amdgcn_wmma_f32_16x16x32_f16(
                       false, a.h, false, b0.h, (short)0, acc0, false, false);
            acc1 = __builtin_amdgcn_wmma_f32_16x16x32_f16(
                       false, a.h, false, b1.h, (short)0, acc1, false, false);
            acc2 = __builtin_amdgcn_wmma_f32_16x16x32_f16(
                       false, a.h, false, b2.h, (short)0, acc2, false, false);
            acc3 = __builtin_amdgcn_wmma_f32_16x16x32_f16(
                       false, a.h, false, b3.h, (short)0, acc3, false, false);
        }
    }

    // ---- store pre-norm y: D layout lane->N, vgpr r -> M = r + khalf*8 ----
    const int x = X0 + col;
    const int obase = m * 16 + khalf * 8;
    #pragma unroll
    for (int r = 0; r < 8; ++r) {
        size_t o = (size_t)n * COUT + obase + r;
        float* orow = out + (o * HUP + (Y0 + rowbase)) * WUP + x;
        orow[0 * WUP] = acc0[r];
        orow[1 * WUP] = acc1[r];
        orow[2 * WUP] = acc2[r];
        orow[3 * WUP] = acc3[r];
    }
}

// ================= kernel 4: per-(n,c) mean / rstd ===========================
__global__ void stats_kernel(const float* __restrict__ y, float* __restrict__ st) {
    const int nc = blockIdx.x;                     // 0..255
    const float* p = y + (size_t)nc * (HUP * WUP);
    float s = 0.0f, s2 = 0.0f;
    for (int i = threadIdx.x; i < HUP * WUP; i += 256) {
        float v = p[i];
        s += v; s2 += v * v;
    }
    __shared__ float sh1[256], sh2[256];
    sh1[threadIdx.x] = s; sh2[threadIdx.x] = s2;
    __syncthreads();
    for (int off = 128; off > 0; off >>= 1) {
        if (threadIdx.x < off) {
            sh1[threadIdx.x] += sh1[threadIdx.x + off];
            sh2[threadIdx.x] += sh2[threadIdx.x + off];
        }
        __syncthreads();
    }
    if (threadIdx.x == 0) {
        const float inv = 1.0f / (float)(HUP * WUP);
        float mean = sh1[0] * inv;
        float var  = sh2[0] * inv - mean * mean;
        st[nc * 2 + 0] = mean;
        st[nc * 2 + 1] = rsqrtf(var + EPS_IN);
    }
}

// ================= kernel 5: in-place instance-norm + LeakyReLU ==============
__global__ void norm_kernel(float* __restrict__ y, const float* __restrict__ st) {
    size_t i = (size_t)blockIdx.x * 256 + threadIdx.x;
    int nc = (int)(i >> 18);                       // / (512*512)
    float mean = st[nc * 2 + 0];
    float rstd = st[nc * 2 + 1];
    float v = (y[i] - mean) * rstd;
    y[i] = (v >= 0.0f) ? v : NEG_SLOPE * v;
}

// ============================== launcher =====================================
extern "C" void kernel_launch(void* const* d_in, const int* in_sizes, int n_in,
                              void* d_out, int out_size, void* d_ws, size_t ws_size,
                              hipStream_t stream) {
    const float* image  = (const float*)d_in[0];   // [8,32,256,256]
    const float* weight = (const float*)d_in[1];   // [32,32,36]
    float* out = (float*)d_out;                    // [8,32,512,512]

    char* ws = (char*)d_ws;
    float*    psi   = (float*)(ws + WS_PSI_OFF);
    _Float16* filt  = (_Float16*)(ws + WS_FILT_OFF);
    float*    stats = (float*)(ws + WS_STAT_OFF);

    // 1) basis
    psi_kernel<<<NBASIS, 256, 0, stream>>>(psi);
    // 2) fold weight x psi -> dense f16 filter (L2-resident, 450 KB)
    fold_kernel<<<(NTAP * COUT * CIN + 255) / 256, 256, 0, stream>>>(weight, psi, filt);
    // 3) fused upsample + conv via WMMA f16->f32 (compute-bound core)
    disco_conv_kernel<<<dim3(WUP / 16, HUP / 16, NB), 256, 0, stream>>>(
        image, (const unsigned*)filt, out);
    // 4) instance-norm stats
    stats_kernel<<<NB * COUT, 256, 0, stream>>>(out, stats);
    // 5) normalize + LeakyReLU in place
    norm_kernel<<<(unsigned)((size_t)NB * COUT * HUP * WUP / 256), 256, 0, stream>>>(out, stats);
}
